// RouterLayer_38439957300070
// MI455X (gfx1250) — matrix operand order repaired and checked by reading, not compiled, not run on hardware
//
#include <hip/hip_runtime.h>
#include <hip/hip_bf16.h>
#include <math.h>

// Problem dims (match reference)
#define B_   8
#define S_   1024
#define H_   1024
#define NH_  16
#define DH_  64
#define DFF_ 4096
#define RB_  128
#define NW_  8

typedef __attribute__((ext_vector_type(16))) __bf16 v16bf;
typedef __attribute__((ext_vector_type(8)))  __bf16 v8bf;
typedef __attribute__((ext_vector_type(8)))  float  v8f;
typedef __attribute__((ext_vector_type(4)))  unsigned int u32x4;
typedef __attribute__((ext_vector_type(8)))  int i32x8;
typedef __attribute__((ext_vector_type(4)))  int i32x4;

static __device__ __forceinline__ float gelu_f(float x) {
  return 0.5f * x * (1.0f + erff(x * 0.70710678118654752f));
}

static __device__ __forceinline__ v16bf cat8(v8bf lo, v8bf hi) {
  return __builtin_shufflevector(lo, hi, 0, 1, 2, 3, 4, 5, 6, 7, 8, 9, 10, 11,
                                 12, 13, 14, 15);
}

// ---------------------------------------------------------------------------
// TDM: issue a 2-D tensor_load_to_lds (one instruction per wave; EXEC ignored)
// Descriptor packing per CDNA5 ISA 8.3/8.4 (D# group0/group1), bf16 elements.
//   tile = tile1 rows x tile0 elements, row stride = stride0 elements,
//   LDS padding: pad_amt_code DWORDs after every pad_int_code chunk.
// This toolchain's builtin is the 6-arg form (extra int32x8 group + cpol).
// ---------------------------------------------------------------------------
static __device__ __forceinline__ void tdm_load_2d(
    unsigned lds_off, const void* gaddr, int tdim0, int tdim1,
    long long stride0, int tile0, int tile1, int pad_int_code,
    int pad_amt_code) {
  unsigned long long ga = (unsigned long long)(uintptr_t)gaddr;
  u32x4 g0;
  g0[0] = 1u;                                              // count=1 (user D#)
  g0[1] = lds_off;                                         // LDS byte address
  g0[2] = (unsigned)ga;                                    // global addr lo
  g0[3] = (unsigned)((ga >> 32) & 0x01FFFFFFu) | (2u << 30);  // hi + type=2
  i32x8 g1;
  g1[0] = (1 << 16)                  // data_size = 2 bytes
        | (1 << 20)                  // pad_enable
        | (pad_int_code << 22)       // DWORDs before padding
        | (pad_amt_code << 25);      // padding DWORDs
  g1[1] = (tdim0 & 0xFFFF) << 16;                              // tensor_dim0 lo
  g1[2] = ((tdim0 >> 16) & 0xFFFF) | ((tdim1 & 0xFFFF) << 16); // dim0 hi|dim1 lo
  g1[3] = ((tdim1 >> 16) & 0xFFFF) | ((tile0 & 0xFFFF) << 16); // dim1 hi|tile0
  g1[4] = tile1 & 0xFFFF;                                      // tile1 (tile2=0)
  g1[5] = (int)(stride0 & 0xFFFFFFFF);                         // dim0 stride lo
  g1[6] = (int)((stride0 >> 32) & 0xFFFF);                     // dim0 stride hi
  g1[7] = 0;
  i32x4 g2 = {0, 0, 0, 0};
  i32x4 g3 = {0, 0, 0, 0};
  i32x8 g4 = {0, 0, 0, 0, 0, 0, 0, 0};
  __builtin_amdgcn_tensor_load_to_lds(g0, g1, g2, g3, g4, 0);
}

// ---------------------------------------------------------------------------
// Transpose + f32->bf16: out[n*K + k] = (bf16)in[k*N + n].  32x32 LDS tiles.
// ---------------------------------------------------------------------------
__global__ __launch_bounds__(256)
void transpose_bf16_kernel(const float* __restrict__ in, __bf16* __restrict__ out,
                           int K, int N) {
  __shared__ float tile[32][33];
  const int n0 = blockIdx.x * 32, k0 = blockIdx.y * 32;
  const int tx = threadIdx.x & 31, ty = threadIdx.x >> 5;  // ty: 0..7
#pragma unroll
  for (int i = 0; i < 32; i += 8)
    tile[ty + i][tx] = in[(size_t)(k0 + ty + i) * N + n0 + tx];
  __syncthreads();
#pragma unroll
  for (int i = 0; i < 32; i += 8)
    out[(size_t)(n0 + ty + i) * K + k0 + tx] = (__bf16)tile[tx][ty + i];
}

// ---------------------------------------------------------------------------
// LayerNorm over H=1024, one workgroup (256 thr) per row, bf16 output
// ---------------------------------------------------------------------------
__global__ __launch_bounds__(256)
void layernorm_bf16_kernel(const float* __restrict__ x,
                           const float* __restrict__ g,
                           const float* __restrict__ b,
                           __bf16* __restrict__ out) {
  const int row = blockIdx.x;
  const int t = threadIdx.x;
  const float* xr = x + (size_t)row * H_;
  float v[4];
  float s = 0.f;
#pragma unroll
  for (int j = 0; j < 4; ++j) { v[j] = xr[t * 4 + j]; s += v[j]; }
  __shared__ float red[256];
  red[t] = s;
  __syncthreads();
  for (int o = 128; o > 0; o >>= 1) {
    if (t < o) red[t] += red[t + o];
    __syncthreads();
  }
  const float mu = red[0] * (1.0f / H_);
  __syncthreads();
  float s2 = 0.f;
#pragma unroll
  for (int j = 0; j < 4; ++j) { float d = v[j] - mu; s2 += d * d; }
  red[t] = s2;
  __syncthreads();
  for (int o = 128; o > 0; o >>= 1) {
    if (t < o) red[t] += red[t + o];
    __syncthreads();
  }
  const float rstd = rsqrtf(red[0] * (1.0f / H_) + 1e-5f);
  __bf16* orow = out + (size_t)row * H_;
#pragma unroll
  for (int j = 0; j < 4; ++j) {
    int c = t * 4 + j;
    orow[c] = (__bf16)((v[j] - mu) * rstd * g[c] + b[c]);
  }
}

// ---------------------------------------------------------------------------
// BF16 WMMA GEMM with transposed weights: out = act(A[M,K] @ Wt[N,K]^T + bias)
// Block tile 128x128, K-step 32; TDM stages both tiles into LDS (wave 0
// issues tensor_load_to_lds + s_wait_tensorcnt, barrier releases the WG).
// 8 waves, each 32x64 via 2 A-frags x 4 B-frags = 8 v_wmma per K-step.
// OUTMODE: 0 = f32 row-major (+residual opt), 1 = bf16 row-major,
//          2 = bf16 transposed per batch: out[(b*H + col)*S + s]  (for V)
// ---------------------------------------------------------------------------
#define GM_BLK 128
#define GN_BLK 128
#define GK_BLK 32
#define TS 40  // LDS row stride in elems (64B data + 16B TDM pad)

template <int ACT, bool RES, int OUTMODE>
__global__ __launch_bounds__(256)
void gemm_bf16_kernel(const __bf16* __restrict__ A, const __bf16* __restrict__ Wt,
                      const float* __restrict__ bias,
                      const float* __restrict__ residual,
                      float* __restrict__ outF, __bf16* __restrict__ outB,
                      int M, int N, int K) {
  __shared__ __attribute__((aligned(16))) __bf16 As[GM_BLK * TS];
  __shared__ __attribute__((aligned(16))) __bf16 Ws[GN_BLK * TS];
  const int t = threadIdx.x;
  const int lane = t & 31;
  const int wave = t >> 5;
  const int mw = wave & 3;   // wave M sub-tile (32 rows)
  const int nw = wave >> 2;  // wave N sub-tile (64 cols)
  const int m0 = blockIdx.y * GM_BLK;
  const int n0 = blockIdx.x * GN_BLK;
  const unsigned as_off = (unsigned)(uintptr_t)&As[0];
  const unsigned ws_off = (unsigned)(uintptr_t)&Ws[0];

  v8f acc[2][4];
#pragma unroll
  for (int a = 0; a < 2; ++a)
#pragma unroll
    for (int i = 0; i < 4; ++i)
#pragma unroll
      for (int j = 0; j < 8; ++j) acc[a][i][j] = 0.f;

  const int akb = (lane < 16) ? 0 : 8;    // A-frag K split base
  const int bkb = (lane < 16) ? 0 : 8;    // B-frag K split base
  const int cl = lane & 15;
  const int half8 = (lane < 16) ? 0 : 8;

  for (int k0 = 0; k0 < K; k0 += GK_BLK) {
    if (t < 32) {  // wave 0 drives the Tensor Data Mover
      // 128x32 bf16 tiles; rows 64B + 16B pad -> LDS stride 80B (TS=40)
      tdm_load_2d(as_off, &A[(size_t)m0 * K + k0], K, M, K, GK_BLK, GM_BLK,
                  /*pad_int=*/3 /*16 DW = 64B*/, /*pad_amt=*/3 /*4 DW = 16B*/);
      tdm_load_2d(ws_off, &Wt[(size_t)n0 * K + k0], K, N, K, GK_BLK, GN_BLK,
                  3, 3);
      __builtin_amdgcn_s_wait_tensorcnt(0);
    }
    __syncthreads();

    v16bf af[2];
#pragma unroll
    for (int am = 0; am < 2; ++am) {
      const __bf16* ap = &As[(mw * 32 + am * 16 + cl) * TS + akb];
      af[am] = cat8(*(const v8bf*)ap, *(const v8bf*)(ap + 16));
    }
#pragma unroll
    for (int tn = 0; tn < 4; ++tn) {
      const __bf16* bp = &Ws[(nw * 64 + tn * 16 + cl) * TS + bkb];
      v16bf bf = cat8(*(const v8bf*)bp, *(const v8bf*)(bp + 16));
#pragma unroll
      for (int am = 0; am < 2; ++am)
        acc[am][tn] = __builtin_amdgcn_wmma_f32_16x16x32_bf16(
            false, af[am], false, bf, (short)0, acc[am][tn], false, false);
    }
    __syncthreads();
  }

  // epilogue
#pragma unroll
  for (int am = 0; am < 2; ++am) {
#pragma unroll
    for (int tn = 0; tn < 4; ++tn) {
      const int col = n0 + nw * 64 + tn * 16 + cl;
      const float bcol = bias[col];
      if (OUTMODE == 2) {
        // transposed bf16 store: rows r are contiguous in out[(b*H+col)*S+s]
        const int row0 = m0 + mw * 32 + am * 16 + half8;  // 8-aligned
        const int bb = row0 >> 10;        // row0 / S_
        const int s0 = row0 & (S_ - 1);   // row0 % S_
        v8bf pack;
#pragma unroll
        for (int r = 0; r < 8; ++r) pack[r] = (__bf16)(acc[am][tn][r] + bcol);
        *(v8bf*)&outB[((size_t)bb * H_ + col) * S_ + s0] = pack;
      } else {
#pragma unroll
        for (int r = 0; r < 8; ++r) {
          const int row = m0 + mw * 32 + am * 16 + r + half8;
          float v = acc[am][tn][r] + bcol;
          if (ACT == 1) v = gelu_f(v);
          if (RES) v += residual[(size_t)row * N + col];
          if (OUTMODE == 1) outB[(size_t)row * N + col] = (__bf16)v;
          else              outF[(size_t)row * N + col] = v;
        }
      }
    }
  }
}

// ---------------------------------------------------------------------------
// Flash attention: grid (S/128, NH, B), 8 waves x 16 query rows per WG.
// K row-major (d-contiguous) and V transposed (key-contiguous); tiles staged
// by the TDM, so all WMMA fragments are contiguous 16B LDS loads.
// ---------------------------------------------------------------------------
#define KS 72  // K-tile LDS stride (128B data + 16B pad)
#define VS 40  // Vt-tile LDS stride (64B data + 16B pad)

__global__ __launch_bounds__(256)
void attention_kernel(const __bf16* __restrict__ Q, const __bf16* __restrict__ Kd,
                      const __bf16* __restrict__ Vt, __bf16* __restrict__ ctx) {
  __shared__ __attribute__((aligned(16))) __bf16 Ks[32 * KS];   // [key][d]
  __shared__ __attribute__((aligned(16))) __bf16 Vs[64 * VS];   // [d][key]
  __shared__ __attribute__((aligned(16))) __bf16 Ps[8][16 * 32];

  const int t = threadIdx.x, lane = t & 31, wave = t >> 5;
  const int qb = blockIdx.x, head = blockIdx.y, b = blockIdx.z;
  const size_t rowbase = (size_t)b * S_;
  const int hcol = head * DH_;
  const int q0 = qb * 128 + wave * 16;
  const int arow = lane & 15;
  const int akb = (lane < 16) ? 0 : 8;
  const int bkb = (lane < 16) ? 0 : 8;
  const int cl = lane & 15;
  const int half8 = (lane < 16) ? 0 : 8;
  const unsigned ks_off = (unsigned)(uintptr_t)&Ks[0];
  const unsigned vs_off = (unsigned)(uintptr_t)&Vs[0];

  // Q fragments (2 K-chunks of 32 over DH=64) straight from global
  v16bf qf[2];
#pragma unroll
  for (int c = 0; c < 2; ++c) {
    const __bf16* qp = Q + (rowbase + q0 + arow) * H_ + hcol + c * 32 + akb;
    __builtin_prefetch(qp, 0, 0);
    qf[c] = cat8(*(const v8bf*)qp, *(const v8bf*)(qp + 16));
  }

  float mi[8], li[8];
  v8f oacc[4];
#pragma unroll
  for (int r = 0; r < 8; ++r) { mi[r] = -1e30f; li[r] = 0.f; }
#pragma unroll
  for (int i = 0; i < 4; ++i)
#pragma unroll
    for (int j = 0; j < 8; ++j) oacc[i][j] = 0.f;

  for (int kb0 = 0; kb0 < S_; kb0 += 32) {
    if (t < 32) {  // wave 0 drives the TDM for both tiles
      // K tile: 32 keys x 64 d (128B rows + 16B pad -> stride 144B)
      tdm_load_2d(ks_off, &Kd[(rowbase + kb0) * H_ + hcol], H_ * S_, S_, H_,
                  64, 32, /*pad_int=*/4 /*32 DW = 128B*/, /*pad_amt=*/3);
      // Vt tile: 64 d x 32 keys (64B rows + 16B pad -> stride 80B)
      tdm_load_2d(vs_off, &Vt[((size_t)b * H_ + hcol) * S_ + kb0], S_ * H_,
                  H_, S_, 32, 64, /*pad_int=*/3, /*pad_amt=*/3);
      __builtin_amdgcn_s_wait_tensorcnt(0);
    }
    __syncthreads();

    // scores: two 16(q)x16(k) tiles, contraction over DH in 2 chunks
    v8f sc[2];
#pragma unroll
    for (int g = 0; g < 2; ++g) {
#pragma unroll
      for (int j = 0; j < 8; ++j) sc[g][j] = 0.f;
#pragma unroll
      for (int c = 0; c < 2; ++c) {
        const __bf16* kp = &Ks[(g * 16 + cl) * KS + c * 32 + bkb];
        v16bf kf = cat8(*(const v8bf*)kp, *(const v8bf*)(kp + 16));
        sc[g] = __builtin_amdgcn_wmma_f32_16x16x32_bf16(
            false, qf[c], false, kf, (short)0, sc[g], false, false);
      }
    }

    // online softmax over this 32-key block (rows live in 16-lane halves)
#pragma unroll
    for (int r = 0; r < 8; ++r) {
      float s0 = sc[0][r] * 0.125f;
      float s1 = sc[1][r] * 0.125f;
      float m = fmaxf(s0, s1);
#pragma unroll
      for (int o = 1; o < 16; o <<= 1) m = fmaxf(m, __shfl_xor(m, o, 32));
      float mn = fmaxf(mi[r], m);
      float alpha = __expf(mi[r] - mn);
      float p0 = __expf(s0 - mn);
      float p1 = __expf(s1 - mn);
      float ls = p0 + p1;
#pragma unroll
      for (int o = 1; o < 16; o <<= 1) ls += __shfl_xor(ls, o, 32);
      li[r] = li[r] * alpha + ls;
      mi[r] = mn;
#pragma unroll
      for (int c4 = 0; c4 < 4; ++c4) oacc[c4][r] *= alpha;
      Ps[wave][(r + half8) * 32 + cl] = (__bf16)p0;
      Ps[wave][(r + half8) * 32 + 16 + cl] = (__bf16)p1;
    }

    // P as A-fragment (same-wave LDS ops are in-order)
    const __bf16* pp = &Ps[wave][arow * 32 + akb];
    v16bf pf = cat8(*(const v8bf*)pp, *(const v8bf*)(pp + 16));
#pragma unroll
    for (int c4 = 0; c4 < 4; ++c4) {
      const __bf16* vp = &Vs[(c4 * 16 + cl) * VS + bkb];
      v16bf vf = cat8(*(const v8bf*)vp, *(const v8bf*)(vp + 16));
      oacc[c4] = __builtin_amdgcn_wmma_f32_16x16x32_bf16(
          false, pf, false, vf, (short)0, oacc[c4], false, false);
    }
    __syncthreads();
  }

  // normalize + store ctx (bf16, row-major)
#pragma unroll
  for (int c4 = 0; c4 < 4; ++c4) {
#pragma unroll
    for (int r = 0; r < 8; ++r) {
      const int row = q0 + r + half8;
      ctx[(rowbase + row) * H_ + hcol + c4 * 16 + cl] =
          (__bf16)(oacc[c4][r] / li[r]);
    }
  }
}

// ---------------------------------------------------------------------------
// Router: pooled mean -> MLP -> logits + argmax. One WG per batch (tiny).
// ---------------------------------------------------------------------------
__global__ __launch_bounds__(256)
void router_kernel(const float* __restrict__ x, const float* __restrict__ rw1,
                   const float* __restrict__ rb1, const float* __restrict__ rw2,
                   const float* __restrict__ rb2, float* __restrict__ logits_out,
                   int* __restrict__ dec) {
  const int b = blockIdx.x;
  const int t = threadIdx.x;
  __shared__ float pooled[H_];
  __shared__ float hidden[RB_];
  __shared__ float lacc[3];
  if (t < 3) lacc[t] = 0.f;
  for (int w = 0; w < NW_; ++w) {
#pragma unroll
    for (int j = 0; j < 4; ++j) {
      const int c = t * 4 + j;
      float s = 0.f;
      for (int tok = 0; tok < 128; ++tok)
        s += x[((size_t)b * S_ + w * 128 + tok) * H_ + c];
      pooled[c] = s * (1.0f / 128.0f);
    }
    __syncthreads();
    if (t < RB_) {
      float a = 0.f;
      for (int hh = 0; hh < H_; ++hh) a += pooled[hh] * rw1[hh * RB_ + t];
      hidden[t] = gelu_f(a + rb1[t]);
    }
    __syncthreads();
    if (t == 0) {
      for (int j = 0; j < 3; ++j) {
        float s = 0.f;
        for (int tt = 0; tt < RB_; ++tt) s += hidden[tt] * rw2[tt * 3 + j];
        lacc[j] += s + rb2[j];
      }
    }
    __syncthreads();
  }
  if (t == 0) {
    float l0 = lacc[0] * (1.0f / NW_);
    float l1 = lacc[1] * (1.0f / NW_);
    float l2 = lacc[2] * (1.0f / NW_);
    logits_out[b * 3 + 0] = l0;
    logits_out[b * 3 + 1] = l1;
    logits_out[b * 3 + 2] = l2;
    int best = 0;
    float bv = l0;
    if (l1 > bv) { best = 1; bv = l1; }
    if (l2 > bv) { best = 2; }
    dec[b] = best;
  }
}

// ---------------------------------------------------------------------------
// Final select: out[b] = {inputs, out_exec, out_rep}[dec[b]]  (float4)
// ---------------------------------------------------------------------------
__global__ __launch_bounds__(256)
void select_kernel(const float* __restrict__ x0, const float* __restrict__ x1,
                   const float* __restrict__ x2, const int* __restrict__ dec,
                   float* __restrict__ out) {
  const size_t i = (size_t)blockIdx.x * blockDim.x + threadIdx.x;
  const int b = (int)((i * 4) / ((size_t)S_ * H_));
  const int d = dec[b];
  const float4* src =
      (d == 0) ? (const float4*)x0 : (d == 1) ? (const float4*)x1 : (const float4*)x2;
  ((float4*)out)[i] = src[i];
}

// ---------------------------------------------------------------------------
// Host orchestration
// ---------------------------------------------------------------------------
extern "C" void kernel_launch(void* const* d_in, const int* in_sizes, int n_in,
                              void* d_out, int out_size, void* d_ws,
                              size_t ws_size, hipStream_t stream) {
  (void)in_sizes; (void)n_in; (void)out_size; (void)ws_size;
  const float* inputs = (const float*)d_in[0];
  const float* ln1_g = (const float*)d_in[1];
  const float* ln1_b = (const float*)d_in[2];
  const float* wq = (const float*)d_in[3];
  const float* bq = (const float*)d_in[4];
  const float* wk = (const float*)d_in[5];
  const float* bk = (const float*)d_in[6];
  const float* wv = (const float*)d_in[7];
  const float* bv = (const float*)d_in[8];
  const float* wo = (const float*)d_in[9];
  const float* bo = (const float*)d_in[10];
  const float* ln2_g = (const float*)d_in[11];
  const float* ln2_b = (const float*)d_in[12];
  const float* wf1 = (const float*)d_in[13];
  const float* bf1 = (const float*)d_in[14];
  const float* wf2 = (const float*)d_in[15];
  const float* bf2 = (const float*)d_in[16];
  const float* rw1 = (const float*)d_in[17];
  const float* rb1 = (const float*)d_in[18];
  const float* rw2 = (const float*)d_in[19];
  const float* rb2 = (const float*)d_in[20];

  const size_t MT = (size_t)B_ * S_;  // 8192 rows

  char* ws = (char*)d_ws;
  size_t off = 0;
  auto alloc = [&](size_t bytes) -> void* {
    void* p = ws + off;
    off = (off + bytes + 255) & ~(size_t)255;
    return p;
  };
  __bf16* wq_t = (__bf16*)alloc((size_t)H_ * H_ * 2);
  __bf16* wk_t = (__bf16*)alloc((size_t)H_ * H_ * 2);
  __bf16* wv_t = (__bf16*)alloc((size_t)H_ * H_ * 2);
  __bf16* wo_t = (__bf16*)alloc((size_t)H_ * H_ * 2);
  __bf16* wf1_t = (__bf16*)alloc((size_t)H_ * DFF_ * 2);
  __bf16* wf2_t = (__bf16*)alloc((size_t)H_ * DFF_ * 2);
  __bf16* hln = (__bf16*)alloc(MT * H_ * 2);
  __bf16* qb = (__bf16*)alloc(MT * H_ * 2);
  __bf16* kb = (__bf16*)alloc(MT * H_ * 2);
  __bf16* vtb = (__bf16*)alloc(MT * H_ * 2);   // V transposed per batch
  __bf16* ctxb = (__bf16*)alloc(MT * H_ * 2);
  __bf16* ffnb = (__bf16*)alloc(MT * DFF_ * 2);
  float* oute = (float*)alloc(MT * H_ * 4);
  float* outr = (float*)alloc(MT * H_ * 4);
  int* dec = (int*)alloc(B_ * 4);

  auto tw = [&](const float* src, __bf16* dst, int K, int N) {
    transpose_bf16_kernel<<<dim3(N / 32, K / 32), 256, 0, stream>>>(src, dst, K, N);
  };
  tw(wq, wq_t, H_, H_);
  tw(wk, wk_t, H_, H_);
  tw(wv, wv_t, H_, H_);
  tw(wo, wo_t, H_, H_);
  tw(wf1, wf1_t, H_, DFF_);
  tw(wf2, wf2_t, DFF_, H_);

  float* logits_out = (float*)d_out + (size_t)B_ * S_ * H_;
  router_kernel<<<dim3(B_), dim3(256), 0, stream>>>(inputs, rw1, rb1, rw2, rb2,
                                                    logits_out, dec);

  auto run_block = [&](const float* Xin, float* Yout) {
    layernorm_bf16_kernel<<<dim3((unsigned)MT), dim3(256), 0, stream>>>(
        Xin, ln1_g, ln1_b, hln);
    dim3 gqkv(H_ / GN_BLK, (unsigned)(MT / GM_BLK));
    gemm_bf16_kernel<0, false, 1><<<gqkv, 256, 0, stream>>>(
        hln, wq_t, bq, nullptr, nullptr, qb, (int)MT, H_, H_);
    gemm_bf16_kernel<0, false, 1><<<gqkv, 256, 0, stream>>>(
        hln, wk_t, bk, nullptr, nullptr, kb, (int)MT, H_, H_);
    gemm_bf16_kernel<0, false, 2><<<gqkv, 256, 0, stream>>>(
        hln, wv_t, bv, nullptr, nullptr, vtb, (int)MT, H_, H_);
    attention_kernel<<<dim3(S_ / 128, NH_, B_), 256, 0, stream>>>(qb, kb, vtb,
                                                                  ctxb);
    gemm_bf16_kernel<0, true, 0><<<gqkv, 256, 0, stream>>>(
        ctxb, wo_t, bo, Xin, Yout, nullptr, (int)MT, H_, H_);
    layernorm_bf16_kernel<<<dim3((unsigned)MT), dim3(256), 0, stream>>>(
        Yout, ln2_g, ln2_b, hln);
    gemm_bf16_kernel<1, false, 1><<<dim3(DFF_ / GN_BLK, (unsigned)(MT / GM_BLK)),
                                    256, 0, stream>>>(
        hln, wf1_t, bf1, nullptr, nullptr, ffnb, (int)MT, DFF_, H_);
    gemm_bf16_kernel<0, true, 0><<<gqkv, 256, 0, stream>>>(
        ffnb, wf2_t, bf2, Yout, Yout, nullptr, (int)MT, H_, DFF_);
  };
  run_block(inputs, oute);
  run_block(oute, outr);

  const size_t nvec4 = (size_t)B_ * S_ * H_ / 4;
  select_kernel<<<dim3((unsigned)(nvec4 / 256)), 256, 0, stream>>>(
      inputs, oute, outr, dec, (float*)d_out);
}
